// predict2feature_CM2FI_41266045780817
// MI455X (gfx1250) — matrix operand order repaired and checked by reading, not compiled, not run on hardware
//
#include <hip/hip_runtime.h>
#include <hip/hip_bf16.h>
#include <math.h>

#define B_ROWS 65536
#define D_IN   1000
#define K_PAD  1024
#define TRUNCK 100
#define OUT_N  526
#define OUT_PAD 528   // 33 * 16
#define LEAKY  0.2f

typedef _Float16 half_t;
typedef _Float16 h8  __attribute__((ext_vector_type(8)));
typedef _Float16 h16 __attribute__((ext_vector_type(16)));
typedef float    f8  __attribute__((ext_vector_type(8)));

// ---------------------------------------------------------------------------
// Pack an [n_rows x n_cols] f32 row-major weight into a zero-padded
// [pad_rows x pad_cols] f16 row-major buffer (K padded so WMMA loops are clean).
// ---------------------------------------------------------------------------
__global__ void pack_w_kernel(const float* __restrict__ W, half_t* __restrict__ Wp,
                              int n_rows, int n_cols, int pad_rows, int pad_cols) {
  int idx = blockIdx.x * blockDim.x + threadIdx.x;
  int total = pad_rows * pad_cols;
  if (idx >= total) return;
  int r = idx / pad_cols;
  int c = idx - r * pad_cols;
  float v = (r < n_rows && c < n_cols) ? W[(size_t)r * n_cols + c] : 0.0f;
  Wp[idx] = (half_t)v;
}

// ---------------------------------------------------------------------------
// Per-row: top-100 via LDS bitonic sort, log/clamp/shift, L2-normalize,
// scatter into a dense f16 row of length K_PAD (zeros elsewhere).
// ---------------------------------------------------------------------------
__global__ __launch_bounds__(256) void topk_scatter_kernel(const float* __restrict__ x,
                                                           half_t* __restrict__ xs16) {
  __shared__ float  sval[K_PAD];
  __shared__ int    sidx[K_PAD];
  __shared__ float  tval[TRUNCK];
  __shared__ float  sred[2];          // [0]=inv_norm  [1]=shift
  __shared__ half_t srow[K_PAD];

  const int row = blockIdx.x;
  const int tid = threadIdx.x;
  const float* xr = x + (size_t)row * D_IN;

  for (int i = tid; i < K_PAD; i += 256) {
    sval[i] = (i < D_IN) ? xr[i] : -INFINITY;
    sidx[i] = i;
  }
  __syncthreads();

  // Bitonic sort, descending by value (pads sink to the tail).
  for (int k = 2; k <= K_PAD; k <<= 1) {
    for (int j = k >> 1; j > 0; j >>= 1) {
      for (int t = tid; t < K_PAD; t += 256) {
        int ixj = t ^ j;
        if (ixj > t) {
          float vi = sval[t], vj = sval[ixj];
          bool descBlock = ((t & k) == 0);
          bool doSwap = descBlock ? (vi < vj) : (vi > vj);
          if (doSwap) {
            sval[t] = vj; sval[ixj] = vi;
            int ti = sidx[t]; sidx[t] = sidx[ixj]; sidx[ixj] = ti;
          }
        }
      }
      __syncthreads();
    }
  }

  // Transform top-100.
  if (tid < TRUNCK) {
    float v = sval[tid];
    tval[tid] = fmaxf(logf(v), -1000.0f) + 50.0f;
  }
  __syncthreads();

  if (tid == 0) {
    float mn = tval[0];
    for (int i = 1; i < TRUNCK; ++i) mn = fminf(mn, tval[i]);
    float shift = fmaxf(-mn, 0.0f);
    float ss = 0.0f;
    for (int i = 0; i < TRUNCK; ++i) { float t = tval[i] + shift; ss += t * t; }
    float norm = sqrtf(ss);
    sred[0] = 1.0f / fmaxf(norm, 1e-12f);
    sred[1] = shift;
  }
  __syncthreads();

  for (int i = tid; i < K_PAD; i += 256) srow[i] = (half_t)0.0f;
  __syncthreads();
  if (tid < TRUNCK) {
    float v = (tval[tid] + sred[1]) * sred[0];
    srow[sidx[tid]] = (half_t)v;
  }
  __syncthreads();

  // Contiguous 128-bit stores of the finished row.
  uint4* d4 = (uint4*)(xs16 + (size_t)row * K_PAD);
  const uint4* s4 = (const uint4*)srow;
  for (int i = tid; i < K_PAD / 8; i += 256) d4[i] = s4[i];
}

// ---------------------------------------------------------------------------
// WMMA fragment loaders (wave32, v_wmma_f32_16x16x32_f16).
// A 16x32 f16: lane m = lane&15; chunks K=[koff,koff+8) and [koff+16,koff+24),
//              koff = (lane>>4)*8  (per ISA 16-bit A table).
// B 32x16 f16 from [N,K] row-major: n = lane&15; same K chunks as A:
//              [koff,koff+8) and [koff+16,koff+24), koff = (lane>>4)*8.
// ---------------------------------------------------------------------------
__device__ __forceinline__ h16 load_frag_a(const half_t* base, int lane, int k0, int ldk) {
  int m    = lane & 15;
  int koff = k0 + ((lane >> 4) << 3);
  h8 lo = *(const h8*)(base + m * ldk + koff);
  h8 hi = *(const h8*)(base + m * ldk + koff + 16);
  return __builtin_shufflevector(lo, hi, 0,1,2,3,4,5,6,7,8,9,10,11,12,13,14,15);
}

__device__ __forceinline__ h16 load_frag_b(const half_t* base, int lane, int k0, int ldk) {
  int n    = lane & 15;
  int koff = k0 + ((lane >> 4) << 3);
  h8 lo = *(const h8*)(base + (size_t)n * ldk + koff);
  h8 hi = *(const h8*)(base + (size_t)n * ldk + koff + 16);
  return __builtin_shufflevector(lo, hi, 0,1,2,3,4,5,6,7,8,9,10,11,12,13,14,15);
}

// ---------------------------------------------------------------------------
// Fused 2-layer MLP. One workgroup = 16 rows, 8 waves.
//  Phase 1: h = LeakyReLU(xs @ W1^T + b1)  -> f16 tile in LDS (never to HBM)
//  Phase 2: out = h @ W2^T + b2            -> f32 to d_out
// ---------------------------------------------------------------------------
__global__ __launch_bounds__(256) void fused_mlp_kernel(const half_t* __restrict__ xs16,
                                                        const half_t* __restrict__ W1p,
                                                        const float*  __restrict__ b1,
                                                        const half_t* __restrict__ W2p,
                                                        const float*  __restrict__ b2,
                                                        float* __restrict__ out) {
  __shared__ half_t xsA[16 * K_PAD];   // 32 KB
  __shared__ half_t hT [16 * K_PAD];   // 32 KB

  const int tid     = threadIdx.x;
  const int lane    = tid & 31;
  const int wave    = tid >> 5;
  const int rowBase = blockIdx.x * 16;

  // Stage the 16xK activation tile into LDS once (shared by all 8 waves).
  {
    const uint4* src = (const uint4*)(xs16 + (size_t)rowBase * K_PAD);
    uint4* dst = (uint4*)xsA;
    for (int i = tid; i < 16 * K_PAD / 8; i += 256) dst[i] = src[i];
  }
  __syncthreads();

  // ---- Phase 1: 64 N-tiles of W1, 8 per wave ----
  for (int tn = wave; tn < K_PAD / 16; tn += 8) {
    f8 acc = {0.f,0.f,0.f,0.f,0.f,0.f,0.f,0.f};
    const half_t* wrow = W1p + (size_t)(tn * 16) * K_PAD;
    for (int k = 0; k < K_PAD; k += 32) {
      h16 a = load_frag_a(xsA, lane, k, K_PAD);
      h16 b = load_frag_b(wrow, lane, k, K_PAD);
      acc = __builtin_amdgcn_wmma_f32_16x16x32_f16(false, a, false, b,
                                                   (short)0, acc, false, false);
    }
    int n = tn * 16 + (lane & 15);
    float bias = (n < D_IN) ? b1[n] : 0.0f;   // padded cols stay exactly 0
    int mBase = (lane >> 4) * 8;              // C/D: VGPR r -> M = r (+8 hi half)
    #pragma unroll
    for (int r = 0; r < 8; ++r) {
      float v = acc[r] + bias;
      v = (v >= 0.0f) ? v : LEAKY * v;
      hT[(mBase + r) * K_PAD + n] = (half_t)v;
    }
  }
  __syncthreads();

  // ---- Phase 2: 33 N-tiles of W2 ----
  for (int tn = wave; tn < OUT_PAD / 16; tn += 8) {
    f8 acc = {0.f,0.f,0.f,0.f,0.f,0.f,0.f,0.f};
    const half_t* wrow = W2p + (size_t)(tn * 16) * K_PAD;
    for (int k = 0; k < K_PAD; k += 32) {
      h16 a = load_frag_a(hT, lane, k, K_PAD);
      h16 b = load_frag_b(wrow, lane, k, K_PAD);
      acc = __builtin_amdgcn_wmma_f32_16x16x32_f16(false, a, false, b,
                                                   (short)0, acc, false, false);
    }
    int j = tn * 16 + (lane & 15);
    if (j < OUT_N) {
      float bias = b2[j];
      int mBase = (lane >> 4) * 8;
      #pragma unroll
      for (int r = 0; r < 8; ++r) {
        int row = rowBase + mBase + r;
        out[(size_t)row * OUT_N + j] = acc[r] + bias;
      }
    }
  }
}

// ---------------------------------------------------------------------------
extern "C" void kernel_launch(void* const* d_in, const int* in_sizes, int n_in,
                              void* d_out, int out_size, void* d_ws, size_t ws_size,
                              hipStream_t stream) {
  (void)in_sizes; (void)n_in; (void)out_size; (void)ws_size;

  const float* x  = (const float*)d_in[0];
  const float* W1 = (const float*)d_in[1];
  const float* b1 = (const float*)d_in[2];
  const float* W2 = (const float*)d_in[3];
  const float* b2 = (const float*)d_in[4];
  float* out = (float*)d_out;

  // Workspace layout (f16): xs16 [B x 1024] | W1p [1024 x 1024] | W2p [528 x 1024]
  half_t* xs16 = (half_t*)d_ws;
  half_t* W1p  = xs16 + (size_t)B_ROWS * K_PAD;
  half_t* W2p  = W1p  + (size_t)K_PAD * K_PAD;

  {
    int total = K_PAD * K_PAD;
    pack_w_kernel<<<(total + 255) / 256, 256, 0, stream>>>(W1, W1p, D_IN, D_IN, K_PAD, K_PAD);
  }
  {
    int total = OUT_PAD * K_PAD;
    pack_w_kernel<<<(total + 255) / 256, 256, 0, stream>>>(W2, W2p, OUT_N, D_IN, OUT_PAD, K_PAD);
  }
  topk_scatter_kernel<<<B_ROWS, 256, 0, stream>>>(x, xs16);
  fused_mlp_kernel<<<B_ROWS / 16, 256, 0, stream>>>(xs16, W1p, b1, W2p, b2, out);
}
